// MultiScaleTemporalBlock_9801115370182
// MI455X (gfx1250) — matrix operand order repaired and checked by reading, not compiled, not run on hardware
//
#include <hip/hip_runtime.h>
#include <hip/hip_bf16.h>
#include <math.h>

// ---------------- problem constants ----------------
#define DIM   1024
#define NHEAD 16
#define DHEAD 64
#define HID   4096
#define WIN   128
#define BATCH 2
#define SEQ   1024
#define NSTR  8                 // 4 modalities * B
#define MROWS (NSTR * SEQ)      // 8192

typedef __attribute__((ext_vector_type(16))) _Float16 v16h;
typedef __attribute__((ext_vector_type(8)))  _Float16 h8v;
typedef __attribute__((ext_vector_type(8)))  float    v8f;

union Frag16 { v16h v; h8v h[2]; _Float16 e[16]; };

static __device__ inline v8f wmma_f16(v16h a, v16h b, v8f c) {
    // (neg_a, A, neg_b, B, c_mod, C, reuse_a, reuse_b)
    return __builtin_amdgcn_wmma_f32_16x16x32_f16(false, a, false, b, (short)0, c,
                                                  false, false);
}

// Load a 16-bit A/B fragment whose two 8-half groups are contiguous in memory.
static __device__ inline v16h load_frag_2x8(const _Float16* p0, const _Float16* p1) {
    Frag16 f;
    f.h[0] = *(const h8v*)p0;
    f.h[1] = *(const h8v*)p1;
    return f.v;
}

// Async 16B copy global -> LDS (ASYNCcnt-tracked, no VGPR round trip).
// Flat addresses in the LDS aperture carry the LDS byte offset in bits[31:0]
// (CDNA5 ISA 10.2), so truncating a generic pointer to a __shared__ object
// yields a valid async-LDS destination.
#define ASYNC_G2L_B128(ldsptr, gptr)                                           \
    asm volatile("global_load_async_to_lds_b128 %0, %1, off"                   \
                 :: "v"((unsigned)(unsigned long long)(ldsptr)),               \
                    "v"((const void*)(gptr))                                   \
                 : "memory")

// ---------------- f32 -> f16 with transpose (weights) ----------------
// in: [K][N] row-major f32, out: [N][K] row-major f16.
__global__ void cvt_transpose_kernel(const float* __restrict__ in,
                                     _Float16* __restrict__ out, int K, int N) {
    int i = blockIdx.x * 256 + threadIdx.x;
    if (i < K * N) {
        int k = i / N, n = i % N;
        out[(size_t)n * K + k] = (_Float16)in[i];
    }
}

// ---------------- LayerNorm (one row per block) -> f16 ----------------
__global__ void ln_kernel(const float* __restrict__ x,
                          const float* __restrict__ g,
                          const float* __restrict__ b,
                          _Float16* __restrict__ out) {
    __shared__ float s1[256], s2[256];
    const int row = blockIdx.x;
    const float* xr = x + (size_t)row * DIM;
    float sum = 0.f, sq = 0.f;
    for (int i = threadIdx.x; i < DIM; i += 256) {
        float v = xr[i];
        sum += v; sq += v * v;
    }
    s1[threadIdx.x] = sum; s2[threadIdx.x] = sq;
    __syncthreads();
    for (int off = 128; off > 0; off >>= 1) {
        if ((int)threadIdx.x < off) {
            s1[threadIdx.x] += s1[threadIdx.x + off];
            s2[threadIdx.x] += s2[threadIdx.x + off];
        }
        __syncthreads();
    }
    const float mu   = s1[0] * (1.f / DIM);
    const float var  = s2[0] * (1.f / DIM) - mu * mu;
    const float rstd = rsqrtf(var + 1e-5f);
    for (int i = threadIdx.x; i < DIM; i += 256)
        out[(size_t)row * DIM + i] = (_Float16)((xr[i] - mu) * rstd * g[i] + b[i]);
}

// ---------------- tiled WMMA GEMM ----------------
// C[M,N] = A16[M,K] @ W[K,N] + bias, W supplied TRANSPOSED as BT16[N][K]
// (weights live in the 192MB L2; B fragments load straight from global as two
//  b128 chunks per lane). A tiles are double-buffered in LDS and filled with
//  global_load_async_to_lds_b128, overlapped with WMMA via s_wait_asynccnt.
// Block: 128 threads (4 waves). Block tile: 128(M) x 64(N), BK = 32.
// Wave w: rows [m0+32w, m0+32w+32) x 64 cols -> 2x4 = 8 wmma per K-step.
#define EPI_F16     0   // out f16 = acc + bias
#define EPI_F32RES  1   // out f32 = acc + bias + residual
#define EPI_SWIGLU  2   // out f16 = silu(aux) * (acc + bias)

template <int EPI>
__global__ __launch_bounds__(128)
void gemm_kernel(const _Float16* __restrict__ A, const _Float16* __restrict__ BT,
                 const float* __restrict__ bias,
                 const _Float16* __restrict__ aux,      // SWIGLU gate input
                 const float* __restrict__ residual,    // F32RES residual
                 float* __restrict__ outF32, _Float16* __restrict__ outF16,
                 int M, int N, int K) {
    __shared__ _Float16 lA[2][128 * 40];   // [buf][row][32 k], row stride 40

    const int tid  = threadIdx.x;
    const int wave = tid >> 5;
    const int lane = tid & 31;
    const int ln16 = lane & 15;
    const int hi   = (lane >> 4) & 1;

    const int n0 = blockIdx.x * 64;
    const int m0 = blockIdx.y * 128;

    v8f acc[2][4];
    const v8f zero = {0.f, 0.f, 0.f, 0.f, 0.f, 0.f, 0.f, 0.f};
#pragma unroll
    for (int mt = 0; mt < 2; ++mt)
#pragma unroll
        for (int nt = 0; nt < 4; ++nt) acc[mt][nt] = zero;

    // per-lane chunk coordinates for async staging: 512 chunks of 8 halves
    // chunk c: row m = c>>2, k-octet k8 = c&3
    const int cm[4]  = { tid >> 2, (tid + 128) >> 2, (tid + 256) >> 2, (tid + 384) >> 2 };
    const int ck8[4] = { tid & 3,  tid & 3,          tid & 3,          tid & 3 };

    // B fragment base pointers (transposed weights, contiguous along K)
    const int khB = hi ? 16 : 0;
    const _Float16* bbase[4];
#pragma unroll
    for (int nt = 0; nt < 4; ++nt)
        bbase[nt] = BT + (size_t)(n0 + nt * 16 + ln16) * K + khB;

    // ---- preload first A tile (k0 = 0) into buffer 0 ----
#pragma unroll
    for (int c = 0; c < 4; ++c)
        ASYNC_G2L_B128(&lA[0][cm[c] * 40 + ck8[c] * 8],
                       A + (size_t)(m0 + cm[c]) * K + ck8[c] * 8);

    const int kbA = hi ? 16 : 0;
    const int nsteps = K >> 5;
    for (int step = 0; step < nsteps; ++step) {
        const int buf = step & 1;
        // issue async fill of the next tile into the other buffer
        if (step + 1 < nsteps) {
            const int kn = (step + 1) << 5;
#pragma unroll
            for (int c = 0; c < 4; ++c)
                ASYNC_G2L_B128(&lA[buf ^ 1][cm[c] * 40 + ck8[c] * 8],
                               A + (size_t)(m0 + cm[c]) * K + kn + ck8[c] * 8);
            asm volatile("s_wait_asynccnt 4" ::: "memory");
        } else {
            asm volatile("s_wait_asynccnt 0" ::: "memory");
        }
        __syncthreads();   // current tile visible to all waves

        const int k0 = step << 5;
        v16h af[2];
#pragma unroll
        for (int mt = 0; mt < 2; ++mt) {
            const _Float16* ap = &lA[buf][(wave * 32 + mt * 16 + ln16) * 40 + kbA];
            af[mt] = load_frag_2x8(ap, ap + 8);
        }
#pragma unroll
        for (int nt = 0; nt < 4; ++nt) {
            const _Float16* bp = bbase[nt] + k0;
            v16h bf = load_frag_2x8(bp, bp + 8);
            acc[0][nt] = wmma_f16(af[0], bf, acc[0][nt]);
            acc[1][nt] = wmma_f16(af[1], bf, acc[1][nt]);
        }
        __syncthreads();   // done reading buf before it is refilled (step+2)
    }

    // epilogue: acc[mt][nt][r] -> row m0+32*wave+16*mt+r+8*hi, col n0+16*nt+ln16
#pragma unroll
    for (int nt = 0; nt < 4; ++nt) {
        const int col = n0 + nt * 16 + ln16;
        const float bv = bias[col];
#pragma unroll
        for (int mt = 0; mt < 2; ++mt)
#pragma unroll
            for (int r = 0; r < 8; ++r) {
                const int row = m0 + wave * 32 + mt * 16 + r + (hi ? 8 : 0);
                const size_t o = (size_t)row * N + col;
                float v = acc[mt][nt][r] + bv;
                if constexpr (EPI == EPI_F16) {
                    outF16[o] = (_Float16)v;
                } else if constexpr (EPI == EPI_F32RES) {
                    outF32[o] = v + residual[o];
                } else {   // SWIGLU: silu(u) * v
                    float u  = (float)aux[o];
                    float su = u / (1.f + expf(-u));
                    outF16[o] = (_Float16)(su * v);
                }
            }
    }
}

// ---------------- banded flash attention ----------------
// qkv f16 [MROWS][3*DIM]; per row: q = [h*64 + d], k at +DIM, v at +2*DIM.
// ctx f16 [MROWS][DIM], head-major columns (h*64 + d).
// One wave per (stream n, head h, 16-query tile). Block = 4 waves.
__global__ __launch_bounds__(128)
void attn_kernel(const _Float16* __restrict__ qkv, _Float16* __restrict__ ctx) {
    __shared__ _Float16 ps[4 * 16 * 40];   // per-wave P scratch [16 q][32 keys]

    const int tid  = threadIdx.x;
    const int wave = tid >> 5;
    const int lane = tid & 31;
    const int ln16 = lane & 15;
    const int hi   = (lane >> 4) & 1;

    const int q0 = (blockIdx.x * 4 + wave) * 16;
    const int h  = blockIdx.y;
    const int n  = blockIdx.z;

    const _Float16* base = qkv + (size_t)n * SEQ * (3 * DIM) + h * DHEAD;
    const float scale = 0.125f;   // 1/sqrt(64)

    // Q fragments (d 0..31 and 32..63)
    const int kbA = hi ? 16 : 0;
    const _Float16* qrow = base + (size_t)(q0 + ln16) * (3 * DIM);
    v16h aq0 = load_frag_2x8(qrow + kbA,      qrow + kbA + 8);
    v16h aq1 = load_frag_2x8(qrow + 32 + kbA, qrow + 32 + kbA + 8);

    float mrow[8], lrow[8];
    v8f o[4];
    const v8f zero = {0.f, 0.f, 0.f, 0.f, 0.f, 0.f, 0.f, 0.f};
#pragma unroll
    for (int r = 0; r < 8; ++r) { mrow[r] = -1e30f; lrow[r] = 0.f; }
#pragma unroll
    for (int t = 0; t < 4; ++t) o[t] = zero;

    _Float16* psw = &ps[wave * 16 * 40];

    const int jlo = (q0 - WIN) > 0 ? (q0 - WIN) : 0;
    const int kb0 = jlo & ~31;
    const int jhi = (q0 + 15 + WIN + 1) < SEQ ? (q0 + 15 + WIN + 1) : SEQ;

    for (int kk = kb0; kk < jhi; kk += 32) {
        // ---- scores for keys [kk, kk+16) and [kk+16, kk+32) ----
        const int khD = hi ? 16 : 0;        // head-dim half per lane group
        const _Float16* k0r = base + (size_t)(kk + ln16)      * (3 * DIM) + DIM;
        const _Float16* k1r = base + (size_t)(kk + 16 + ln16) * (3 * DIM) + DIM;
        v8f s0 = zero, s1 = zero;
        s0 = wmma_f16(aq0, load_frag_2x8(k0r + khD,      k0r + khD + 8),      s0);
        s0 = wmma_f16(aq1, load_frag_2x8(k0r + 32 + khD, k0r + 32 + khD + 8), s0);
        s1 = wmma_f16(aq0, load_frag_2x8(k1r + khD,      k1r + khD + 8),      s1);
        s1 = wmma_f16(aq1, load_frag_2x8(k1r + 32 + khD, k1r + 32 + khD + 8), s1);

        // ---- scale + band mask + online softmax ----
        float v0[8], v1[8], mx[8];
#pragma unroll
        for (int r = 0; r < 8; ++r) {
            const int row = q0 + r + (hi ? 8 : 0);
            const int c0 = kk + ln16, c1 = c0 + 16;
            float a0 = s0[r] * scale, a1 = s1[r] * scale;
            int d0 = row - c0; if (d0 < 0) d0 = -d0;
            int d1 = row - c1; if (d1 < 0) d1 = -d1;
            v0[r] = (d0 <= WIN) ? a0 : -1e30f;
            v1[r] = (d1 <= WIN) ? a1 : -1e30f;
            mx[r] = v0[r] > v1[r] ? v0[r] : v1[r];
        }
#pragma unroll
        for (int off = 1; off < 16; off <<= 1)
#pragma unroll
            for (int r = 0; r < 8; ++r) {
                float t = __shfl_xor(mx[r], off, 32);
                mx[r] = mx[r] > t ? mx[r] : t;
            }

        float rs[8], alpha[8];
#pragma unroll
        for (int r = 0; r < 8; ++r) {
            float mnew = mrow[r] > mx[r] ? mrow[r] : mx[r];
            alpha[r] = expf(mrow[r] - mnew);
            v0[r] = expf(v0[r] - mnew);
            v1[r] = expf(v1[r] - mnew);
            rs[r] = v0[r] + v1[r];
            mrow[r] = mnew;
        }
#pragma unroll
        for (int off = 1; off < 16; off <<= 1)
#pragma unroll
            for (int r = 0; r < 8; ++r)
                rs[r] += __shfl_xor(rs[r], off, 32);
#pragma unroll
        for (int r = 0; r < 8; ++r) lrow[r] = lrow[r] * alpha[r] + rs[r];
#pragma unroll
        for (int t = 0; t < 4; ++t)
#pragma unroll
            for (int r = 0; r < 8; ++r) o[t][r] *= alpha[r];

        // ---- route P through LDS into A-fragment layout ----
#pragma unroll
        for (int r = 0; r < 8; ++r) {
            const int prow = r + (hi ? 8 : 0);
            psw[prow * 40 + ln16]      = (_Float16)v0[r];
            psw[prow * 40 + ln16 + 16] = (_Float16)v1[r];
        }
        asm volatile("s_wait_dscnt 0" ::: "memory");
        const _Float16* pp = &psw[ln16 * 40 + kbA];
    v16h pfrag = load_frag_2x8(pp, pp + 8);

        // ---- P @ V : B fragment = V[key][d], key = kk + kh + j ----
        const int khK = hi ? 16 : 0;
#pragma unroll
        for (int dt = 0; dt < 4; ++dt) {
            const _Float16* vb = base + 2 * DIM + dt * 16 + ln16;
            Frag16 vf;
#pragma unroll
            for (int j = 0; j < 16; ++j)
                vf.e[j] = vb[(size_t)(kk + khK + j) * (3 * DIM)];
            o[dt] = wmma_f16(pfrag, vf.v, o[dt]);
        }
    }

    // ---- normalize + store ctx ----
    float inv[8];
#pragma unroll
    for (int r = 0; r < 8; ++r) inv[r] = 1.f / lrow[r];
#pragma unroll
    for (int dt = 0; dt < 4; ++dt)
#pragma unroll
        for (int r = 0; r < 8; ++r) {
            const int row = q0 + r + (hi ? 8 : 0);
            ctx[((size_t)(n * SEQ + row)) * DIM + h * DHEAD + dt * 16 + ln16] =
                (_Float16)(o[dt][r] * inv[r]);
        }
}

// ---------------- launcher ----------------
extern "C" void kernel_launch(void* const* d_in, const int* in_sizes, int n_in,
                              void* d_out, int out_size, void* d_ws, size_t ws_size,
                              hipStream_t stream) {
    const float* ln1_g = (const float*)d_in[4];
    const float* ln1_b = (const float*)d_in[5];
    const float* Wqkv  = (const float*)d_in[6];
    const float* bqkv  = (const float*)d_in[7];
    const float* Wo    = (const float*)d_in[8];
    const float* bo    = (const float*)d_in[9];
    const float* ln2_g = (const float*)d_in[10];
    const float* ln2_b = (const float*)d_in[11];
    const float* W1    = (const float*)d_in[12];
    const float* b1    = (const float*)d_in[13];
    const float* W2    = (const float*)d_in[14];
    const float* b2    = (const float*)d_in[15];
    const float* W3    = (const float*)d_in[16];
    const float* b3    = (const float*)d_in[17];

    float* x = (float*)d_out;   // residual stream [8, S, D]

    // workspace layout (f16 buffers); weights stored TRANSPOSED [N][K]
    char* ws = (char*)d_ws;
    size_t off = 0;
    _Float16* WqkvT = (_Float16*)(ws + off); off += (size_t)DIM * 3 * DIM * 2;
    _Float16* WoT   = (_Float16*)(ws + off); off += (size_t)DIM * DIM * 2;
    _Float16* W1T   = (_Float16*)(ws + off); off += (size_t)DIM * HID * 2;
    _Float16* W2T   = (_Float16*)(ws + off); off += (size_t)DIM * HID * 2;
    _Float16* W3T   = (_Float16*)(ws + off); off += (size_t)HID * DIM * 2;
    _Float16* h16   = (_Float16*)(ws + off); off += (size_t)MROWS * DIM * 2;
    _Float16* qkv16 = (_Float16*)(ws + off); off += (size_t)MROWS * 3 * DIM * 2;
    _Float16* ctx16 = (_Float16*)(ws + off); off += (size_t)MROWS * DIM * 2;
    _Float16* u16   = (_Float16*)(ws + off); off += (size_t)MROWS * HID * 2;
    _Float16* g16   = (_Float16*)(ws + off); off += (size_t)MROWS * HID * 2;

    // 1) concat [fusion, visual, text, audio] -> x
    const size_t strBytes = (size_t)BATCH * SEQ * DIM * sizeof(float);
    for (int i = 0; i < 4; ++i)
        hipMemcpyAsync((char*)x + i * strBytes, d_in[i], strBytes,
                       hipMemcpyDeviceToDevice, stream);

    // 2) weights -> f16, transposed
    auto cvtT = [&](const float* src, _Float16* dst, int K, int N) {
        int n = K * N;
        cvt_transpose_kernel<<<(n + 255) / 256, 256, 0, stream>>>(src, dst, K, N);
    };
    cvtT(Wqkv, WqkvT, DIM, 3 * DIM);
    cvtT(Wo,   WoT,   DIM, DIM);
    cvtT(W1,   W1T,   DIM, HID);
    cvtT(W2,   W2T,   DIM, HID);
    cvtT(W3,   W3T,   HID, DIM);

    // 3) LN1
    ln_kernel<<<MROWS, 256, 0, stream>>>(x, ln1_g, ln1_b, h16);

    // 4) qkv = h @ Wqkv + bqkv  (f16 out)
    gemm_kernel<EPI_F16><<<dim3(3 * DIM / 64, MROWS / 128), 128, 0, stream>>>(
        h16, WqkvT, bqkv, nullptr, nullptr, nullptr, qkv16, MROWS, 3 * DIM, DIM);

    // 5) banded attention
    attn_kernel<<<dim3(SEQ / 64, NHEAD, NSTR), 128, 0, stream>>>(qkv16, ctx16);

    // 6) x += ctx @ Wo + bo
    gemm_kernel<EPI_F32RES><<<dim3(DIM / 64, MROWS / 128), 128, 0, stream>>>(
        ctx16, WoT, bo, nullptr, x, x, nullptr, MROWS, DIM, DIM);

    // 7) LN2
    ln_kernel<<<MROWS, 256, 0, stream>>>(x, ln2_g, ln2_b, h16);

    // 8) u = h2 @ W1 + b1  (f16)
    gemm_kernel<EPI_F16><<<dim3(HID / 64, MROWS / 128), 128, 0, stream>>>(
        h16, W1T, b1, nullptr, nullptr, nullptr, u16, MROWS, HID, DIM);

    // 9) g = silu(u) * (h2 @ W2 + b2)  (f16)
    gemm_kernel<EPI_SWIGLU><<<dim3(HID / 64, MROWS / 128), 128, 0, stream>>>(
        h16, W2T, b2, u16, nullptr, nullptr, g16, MROWS, HID, DIM);

    // 10) x += g @ W3 + b3
    gemm_kernel<EPI_F32RES><<<dim3(DIM / 64, MROWS / 128), 128, 0, stream>>>(
        g16, W3T, b3, nullptr, x, x, nullptr, MROWS, DIM, HID);

    (void)in_sizes; (void)n_in; (void)out_size; (void)ws_size;
}
